// BIDIRECTIONAL_VRNN_7516192768897
// MI455X (gfx1250) — compile-verified, matrix operand order checked by reading, not run on hardware
//
#include <hip/hip_runtime.h>
#include <math.h>

// ---------------- CDNA5 types ----------------
typedef __bf16 bf16_t;
typedef __bf16 v16bf __attribute__((ext_vector_type(16)));
typedef float  v8f   __attribute__((ext_vector_type(8)));

// Global (AS1) pointer typedefs: force global_load/global_store instead of flat_*
// (FLAT ops consume both LOADcnt and DScnt and tie up the LDS path on CDNA5).
typedef const __attribute__((address_space(1))) v16bf*  g_v16bf_cp;
typedef const __attribute__((address_space(1))) float*  g_f32_cp;
typedef __attribute__((address_space(1))) float*        g_f32_p;
typedef const __attribute__((address_space(1))) bf16_t* g_bf16_cp;
typedef __attribute__((address_space(1))) bf16_t*       g_bf16_p;
typedef const __attribute__((address_space(1))) int*    g_i32_cp;
typedef __attribute__((address_space(1))) unsigned*     g_u32_p;

// ---------------- problem constants ----------------
#define T_   128
#define B_   32
#define E_   256
#define H_   512
#define Z_   128
#define V_   10000
#define H2_  1024
#define CATW 2048          // [phi_x | phi_z | h_f | h_b] concat buffer width
#define EPS_ 1e-8f

// ---------------- kernel config ----------------
#define NWG  64            // persistent workgroups (must be co-resident)
#define TPB  256           // 8 wave32s per WG
#define KC   512           // K-chunk staged in LDS
#define SAP  528           // padded LDS row stride (bf16) -> 1056B rows, 32B aligned, bank-skewed

#define ACT_NONE     0
#define ACT_RELU     1
#define ACT_SIGMOID  2
#define ACT_SOFTPLUS 3

// weight enum (order fixed, matches host tables)
enum { W_PHI1, W_PHI2, W_PHIZ, W_ENC1, W_ENC2, W_ENCM, W_ENCS, W_PRIOR, W_PRM, W_PRS,
       W_DEC1, W_DEC2, W_DECM, W_DECS, W_GIF, W_GHF, W_GIB, W_GHB, W_OUT, NW };
// bias enum
enum { BI_PHI1, BI_PHI2, BI_PHIZ, BI_ENC1, BI_ENC2, BI_ENCM, BI_ENCS, BI_PRIOR, BI_PRM,
       BI_PRS, BI_DEC1, BI_DEC2, BI_DECM, BI_DECS, BI_OUT, NB };
// scratch enum
enum { S_XT, S_T1, S_CAT, S_ET1, S_ENCH, S_ENCM, S_ENCS, S_PRH, S_PRM, S_PRS, S_ZB,
       S_DT1, S_DECH, S_GIF, S_GHF, S_GIB, S_GHB, S_LOGITS, S_RROW, S_ACC, NS };

struct PrepArgs {
  const float* src[NW];
  bf16_t*      dst[NW];
  int          cnt[NW];
  unsigned*    bar;
  float*       acc;
};

struct VArgs {
  const int*    inputs;   // (T,B)
  const float*  eps;      // (T,B,Z)
  const float*  embed;    // (V+1,E)
  const bf16_t* w[NW];
  const float*  bias[NB];
  float*        s[NS];
  unsigned*     bar;
  float*        dout;
};

// ---------------- weight prep: f32 -> bf16 (weights live in L2 thereafter) ----
__global__ void vrnn_prep(PrepArgs a) {
  long tid    = (long)blockIdx.x * blockDim.x + threadIdx.x;
  long stride = (long)gridDim.x * blockDim.x;
  for (int s = 0; s < NW; ++s) {
    g_f32_cp src = (g_f32_cp)a.src[s];
    g_bf16_p dst = (g_bf16_p)a.dst[s];
    int      n   = a.cnt[s];
    for (long i = tid; i < n; i += stride) dst[i] = (bf16_t)src[i];
  }
  if (tid == 0) {
    g_u32_p bar = (g_u32_p)a.bar;
    g_f32_p acc = (g_f32_p)a.acc;
    bar[0] = 0u; bar[1] = 0u;              // grid-barrier count + sense
    acc[0] = 0.f; acc[1] = 0.f;            // kld / recon accumulators
  }
}

// ---------------- grid barrier (sense-reversing, agent scope) ----------------
__device__ __forceinline__ void gridbar(unsigned* cnt, unsigned* sns,
                                        unsigned* lsp, int nwg) {
  __syncthreads();
  if (threadIdx.x == 0) {
    unsigned s = *lsp ^ 1u; *lsp = s;
    unsigned old = __hip_atomic_fetch_add(cnt, 1u, __ATOMIC_ACQ_REL,
                                          __HIP_MEMORY_SCOPE_AGENT);
    if (old == (unsigned)(nwg - 1)) {
      __hip_atomic_store(cnt, 0u, __ATOMIC_RELAXED, __HIP_MEMORY_SCOPE_AGENT);
      __hip_atomic_store(sns, s,  __ATOMIC_RELEASE, __HIP_MEMORY_SCOPE_AGENT);
    } else {
      while (__hip_atomic_load(sns, __ATOMIC_ACQUIRE,
                               __HIP_MEMORY_SCOPE_AGENT) != s)
        __builtin_amdgcn_s_sleep(2);
    }
  }
  __syncthreads();
}

__device__ __forceinline__ float act_apply(float v, int act) {
  if (act == ACT_RELU)     return fmaxf(v, 0.f);
  if (act == ACT_SIGMOID)  return 1.f / (1.f + __expf(-v));
  if (act == ACT_SOFTPLUS) return (v > 20.f) ? v : log1pf(__expf(v));
  return v;
}

// ---------------- WMMA GEMM phase: C(32xN) = act( A(32xK) @ W(NxK)^T + b ) ----
// A: f32 row-major (lda), staged into LDS as bf16 per K-chunk (shared by all 8 waves).
// W: bf16 row-major (ldw).  Each wave owns up to 2 N-tiles of 16 columns and computes
// both 16-row M-tiles per B fragment (register-level B reuse; LDS-level A reuse).
__device__ void gemm_tiles(const float* __restrict__ Aact, int lda,
                           const bf16_t* __restrict__ W, int ldw,
                           const float* __restrict__ bias,
                           float* __restrict__ C, int ldc,
                           float* __restrict__ C2, int ldc2,
                           int K, int N, int act, int accum,
                           bf16_t* sA, int gw, int gwN) {
  const int tid = threadIdx.x;
  const int lane = tid & 31;
  const int hi = lane >> 4;          // lane group: K-half (A) / K-half (B)
  const int ml = lane & 15;          // row (A) / column (B,C)
  const int ntiles = N >> 4;
  const int tiles[2] = { gw, gw + gwN };
  g_f32_cp Ag = (g_f32_cp)Aact;
  v8f acc2[2][2] = {};               // [tile][mtile]

  for (int kb = 0; kb < K; kb += KC) {
    const int kw = (K - kb < KC) ? (K - kb) : KC;      // 128/256/512 (pow2)
    const int lg = (kw == 512) ? 9 : (kw == 256) ? 8 : 7;
    __syncthreads();                                   // protect sA reuse
    for (int i = tid; i < (kw << 5); i += TPB) {       // stage 32 x kw -> bf16 LDS
      int m = i >> lg, k = i & (kw - 1);
      sA[m * SAP + k] = (bf16_t)Ag[(size_t)m * lda + kb + k];
    }
    __syncthreads();
#pragma unroll
    for (int p = 0; p < 2; ++p) {
      if (tiles[p] >= ntiles) continue;
      const bf16_t* wrow = W + (size_t)(tiles[p] * 16 + ml) * ldw + kb + (hi << 4);
      const bf16_t* ar0  = sA + ml * SAP + (hi << 4);
      const bf16_t* ar1  = sA + (ml + 16) * SAP + (hi << 4);
      for (int kk = 0; kk < kw; kk += 32) {
        v16bf bm = *(g_v16bf_cp)(const void*)(wrow + kk); // global_load_b128 x2
        __builtin_prefetch(wrow + kk + 128, 0, 1);        // prefetch_b8 (L2 read-ahead)
        v16bf a0 = *(const v16bf*)(ar0 + kk);             // ds_load_b128 x2
        v16bf a1 = *(const v16bf*)(ar1 + kk);
        acc2[p][0] = __builtin_amdgcn_wmma_f32_16x16x32_bf16(
            false, a0, false, bm, (short)0, acc2[p][0], false, false);
        acc2[p][1] = __builtin_amdgcn_wmma_f32_16x16x32_bf16(
            false, a1, false, bm, (short)0, acc2[p][1], false, false);
      }
    }
  }
  // epilogue: C layout -> lane (hi,ml), vgpr r: m = 8*hi + r, n = ml
  g_f32_p Cg  = (g_f32_p)C;
  g_f32_p C2g = (g_f32_p)C2;
  g_f32_cp bg = (g_f32_cp)bias;
#pragma unroll
  for (int p = 0; p < 2; ++p) {
    if (tiles[p] >= ntiles) continue;
    const int n = tiles[p] * 16 + ml;
    const float bi = bias ? bg[n] : 0.f;
#pragma unroll
    for (int mt = 0; mt < 2; ++mt) {
#pragma unroll
      for (int r = 0; r < 8; ++r) {
        const int m = mt * 16 + (hi << 3) + r;
        float v = acc2[p][mt][r];
        if (accum) v += Cg[(size_t)m * ldc + n];       // split-K second pass
        v += bi;
        v = act_apply(v, act);
        Cg[(size_t)m * ldc + n] = v;
        if (C2) C2g[(size_t)m * ldc2 + n] = v;
      }
    }
  }
}

// ---------------- persistent VRNN kernel ----------------
__global__ __launch_bounds__(TPB) void vrnn_main(VArgs A) {
  __shared__ bf16_t sA[32 * SAP];
  __shared__ float  red[TPB];
  __shared__ unsigned lsense;

  const int tid = threadIdx.x;
  const int wg = blockIdx.x, nwg = gridDim.x;
  const int wid = tid >> 5;
  const int gw  = wg * (TPB / 32) + wid;
  const int gwN = nwg * (TPB / 32);
  if (tid == 0) lsense = 0u;
  __syncthreads();

  unsigned* cnt = A.bar;
  unsigned* sns = A.bar + 1;
  float* cat = A.s[S_CAT];
  float* out_encm = A.dout + 2;
  float* out_encs = out_encm + (size_t)T_ * B_ * Z_;
  float* out_decm = out_encs + (size_t)T_ * B_ * Z_;
  float* out_decs = out_decm + (size_t)T_ * B_ * E_;

  g_i32_cp inputs = (g_i32_cp)A.inputs;
  g_f32_cp embed  = (g_f32_cp)A.embed;
  g_f32_cp epsg   = (g_f32_cp)A.eps;
  g_f32_p  catg   = (g_f32_p)cat;
  g_f32_p  xt     = (g_f32_p)A.s[S_XT];
  g_f32_p  zb     = (g_f32_p)A.s[S_ZB];
  g_f32_cp encm_g = (g_f32_cp)A.s[S_ENCM];
  g_f32_cp encs_g = (g_f32_cp)A.s[S_ENCS];
  g_f32_cp prm_g  = (g_f32_cp)A.s[S_PRM];
  g_f32_cp prs_g  = (g_f32_cp)A.s[S_PRS];
  g_f32_p  acc_g  = (g_f32_p)A.s[S_ACC];
  g_f32_p  rrow_g = (g_f32_p)A.s[S_RROW];

  // init: h0 = 0 (zero whole concat buffer)
  for (int i = wg * TPB + tid; i < B_ * CATW; i += nwg * TPB) catg[i] = 0.f;
  gridbar(cnt, sns, &lsense, nwg);

  for (int t = 0; t < T_; ++t) {
    // ---- P0: embedding gather x_t ----
    for (int i = wg * TPB + tid; i < B_ * E_; i += nwg * TPB) {
      int b = i >> 8, e = i & 255;
      int tok = inputs[t * B_ + b];
      xt[i] = embed[(size_t)tok * E_ + e];
    }
    gridbar(cnt, sns, &lsense, nwg);

    // ---- P1: phi_x layer1 ----
    gemm_tiles(A.s[S_XT], E_, A.w[W_PHI1], E_, A.bias[BI_PHI1],
               A.s[S_T1], H_, nullptr, 0, E_, H_, ACT_RELU, 0, sA, gw, gwN);
    gridbar(cnt, sns, &lsense, nwg);

    // ---- P2: phi_x layer2 -> cat[0:512) ----
    gemm_tiles(A.s[S_T1], H_, A.w[W_PHI2], H_, A.bias[BI_PHI2],
               cat + 0, CATW, nullptr, 0, H_, H_, ACT_RELU, 0, sA, gw, gwN);
    gridbar(cnt, sns, &lsense, nwg);

    // ---- P3: enc layer1 (split-K over [phi_x | h_comb]) + prior hidden ----
    gemm_tiles(cat + 0, CATW, A.w[W_ENC1], H_ + H2_, nullptr,
               A.s[S_ET1], H2_, nullptr, 0, H_, H2_, ACT_NONE, 0, sA, gw, gwN);
    gemm_tiles(cat + 2 * H_, CATW, A.w[W_ENC1] + H_, H_ + H2_, A.bias[BI_ENC1],
               A.s[S_ET1], H2_, nullptr, 0, H2_, H2_, ACT_RELU, 1, sA, gw, gwN);
    gemm_tiles(cat + 2 * H_, CATW, A.w[W_PRIOR], H2_, A.bias[BI_PRIOR],
               A.s[S_PRH], H2_, nullptr, 0, H2_, H2_, ACT_RELU, 0, sA, gw, gwN);
    gridbar(cnt, sns, &lsense, nwg);

    // ---- P4: enc layer2, prior mean/std ----
    gemm_tiles(A.s[S_ET1], H2_, A.w[W_ENC2], H2_, A.bias[BI_ENC2],
               A.s[S_ENCH], H2_, nullptr, 0, H2_, H2_, ACT_RELU, 0, sA, gw, gwN);
    gemm_tiles(A.s[S_PRH], H2_, A.w[W_PRM], H2_, A.bias[BI_PRM],
               A.s[S_PRM], Z_, nullptr, 0, H2_, Z_, ACT_NONE, 0, sA, gw, gwN);
    gemm_tiles(A.s[S_PRH], H2_, A.w[W_PRS], H2_, A.bias[BI_PRS],
               A.s[S_PRS], Z_, nullptr, 0, H2_, Z_, ACT_SOFTPLUS, 0, sA, gw, gwN);
    gridbar(cnt, sns, &lsense, nwg);

    // ---- P5: enc mean/std (also streamed to output) ----
    gemm_tiles(A.s[S_ENCH], H2_, A.w[W_ENCM], H2_, A.bias[BI_ENCM],
               A.s[S_ENCM], Z_, out_encm + (size_t)t * B_ * Z_, Z_,
               H2_, Z_, ACT_NONE, 0, sA, gw, gwN);
    gemm_tiles(A.s[S_ENCH], H2_, A.w[W_ENCS], H2_, A.bias[BI_ENCS],
               A.s[S_ENCS], Z_, out_encs + (size_t)t * B_ * Z_, Z_,
               H2_, Z_, ACT_SOFTPLUS, 0, sA, gw, gwN);
    gridbar(cnt, sns, &lsense, nwg);

    // ---- P6: z = eps*std+mean ; wg0 also does deterministic KLD reduction ----
    for (int i = wg * TPB + tid; i < B_ * Z_; i += nwg * TPB)
      zb[i] = epsg[(size_t)t * B_ * Z_ + i] * encs_g[i] + encm_g[i];
    if (wg == 0) {
      float part = 0.f;
      for (int i = tid; i < B_ * Z_; i += TPB) {
        float es = encs_g[i], em = encm_g[i];
        float ps = prs_g[i],  pm = prm_g[i];
        float d = em - pm;
        part += 2.f * __logf(ps + EPS_) - 2.f * __logf(es + EPS_)
              + (es * es + d * d) / (ps * ps) - 1.f;
      }
      red[tid] = part; __syncthreads();
      for (int s = TPB / 2; s > 0; s >>= 1) {
        if (tid < s) red[tid] += red[tid + s];
        __syncthreads();
      }
      if (tid == 0) acc_g[0] += 0.5f * red[0] / (float)(B_ * Z_);
    }
    gridbar(cnt, sns, &lsense, nwg);

    // ---- P7: phi_z -> cat[512:1024) ----
    gemm_tiles(A.s[S_ZB], Z_, A.w[W_PHIZ], Z_, A.bias[BI_PHIZ],
               cat + H_, CATW, nullptr, 0, Z_, H_, ACT_RELU, 0, sA, gw, gwN);
    gridbar(cnt, sns, &lsense, nwg);

    // ---- P8: dec layer1 ([phi_z|h] contiguous) + GRU input/hidden GEMMs ----
    gemm_tiles(cat + H_, CATW, A.w[W_DEC1], H_ + H2_, A.bias[BI_DEC1],
               A.s[S_DT1], H2_, nullptr, 0, H_ + H2_, H2_, ACT_RELU, 0, sA, gw, gwN);
    gemm_tiles(cat + 0, CATW, A.w[W_GIF], 2 * H_, nullptr,
               A.s[S_GIF], 3 * H_, nullptr, 0, 2 * H_, 3 * H_, ACT_NONE, 0, sA, gw, gwN);
    gemm_tiles(cat + 0, CATW, A.w[W_GIB], 2 * H_, nullptr,
               A.s[S_GIB], 3 * H_, nullptr, 0, 2 * H_, 3 * H_, ACT_NONE, 0, sA, gw, gwN);
    gemm_tiles(cat + 2 * H_, CATW, A.w[W_GHF], H_, nullptr,
               A.s[S_GHF], 3 * H_, nullptr, 0, H_, 3 * H_, ACT_NONE, 0, sA, gw, gwN);
    gemm_tiles(cat + 3 * H_, CATW, A.w[W_GHB], H_, nullptr,
               A.s[S_GHB], 3 * H_, nullptr, 0, H_, 3 * H_, ACT_NONE, 0, sA, gw, gwN);
    gridbar(cnt, sns, &lsense, nwg);

    // ---- P9: dec layer2 + GRU gate math (updates h in-place; old h no longer read) ----
    gemm_tiles(A.s[S_DT1], H2_, A.w[W_DEC2], H2_, A.bias[BI_DEC2],
               A.s[S_DECH], H2_, nullptr, 0, H2_, H2_, ACT_RELU, 0, sA, gw, gwN);
    {
      g_f32_cp gif_g = (g_f32_cp)A.s[S_GIF];
      g_f32_cp ghf_g = (g_f32_cp)A.s[S_GHF];
      g_f32_cp gib_g = (g_f32_cp)A.s[S_GIB];
      g_f32_cp ghb_g = (g_f32_cp)A.s[S_GHB];
      for (int i = wg * TPB + tid; i < B_ * H_; i += nwg * TPB) {
        int m = i >> 9, j = i & (H_ - 1);
        size_t base = (size_t)m * 3 * H_;
        float hf = catg[m * CATW + 2 * H_ + j];
        float r  = 1.f / (1.f + __expf(-(gif_g[base + j] + ghf_g[base + j])));
        float zg = 1.f / (1.f + __expf(-(gif_g[base + H_ + j] + ghf_g[base + H_ + j])));
        float nn = tanhf(gif_g[base + 2 * H_ + j] + r * ghf_g[base + 2 * H_ + j]);
        catg[m * CATW + 2 * H_ + j] = (1.f - zg) * nn + zg * hf;
        float hb = catg[m * CATW + 3 * H_ + j];
        r  = 1.f / (1.f + __expf(-(gib_g[base + j] + ghb_g[base + j])));
        zg = 1.f / (1.f + __expf(-(gib_g[base + H_ + j] + ghb_g[base + H_ + j])));
        nn = tanhf(gib_g[base + 2 * H_ + j] + r * ghb_g[base + 2 * H_ + j]);
        catg[m * CATW + 3 * H_ + j] = (1.f - zg) * nn + zg * hb;
      }
    }
    gridbar(cnt, sns, &lsense, nwg);

    // ---- P10: dec mean/std (straight to output) + logits (big N=10000 GEMM) ----
    gemm_tiles(A.s[S_DECH], H2_, A.w[W_DECM], H2_, A.bias[BI_DECM],
               out_decm + (size_t)t * B_ * E_, E_, nullptr, 0,
               H2_, E_, ACT_SIGMOID, 0, sA, gw, gwN);
    gemm_tiles(A.s[S_DECH], H2_, A.w[W_DECS], H2_, A.bias[BI_DECS],
               out_decs + (size_t)t * B_ * E_, E_, nullptr, 0,
               H2_, E_, ACT_SOFTPLUS, 0, sA, gw, gwN);
    gemm_tiles(A.s[S_DECH], H2_, A.w[W_OUT], H2_, A.bias[BI_OUT],
               A.s[S_LOGITS], V_, nullptr, 0, H2_, V_, ACT_NONE, 0, sA, gw, gwN);
    gridbar(cnt, sns, &lsense, nwg);

    // ---- P11: per-row log-softmax / recon (wg b handles batch row b; deterministic) ----
    if (wg < B_) {
      g_f32_cp row = (g_f32_cp)(A.s[S_LOGITS] + (size_t)wg * V_);
      float mx = -3.4e38f;
      for (int i = tid; i < V_; i += TPB) mx = fmaxf(mx, row[i]);
      red[tid] = mx; __syncthreads();
      for (int s = TPB / 2; s > 0; s >>= 1) {
        if (tid < s) red[tid] = fmaxf(red[tid], red[tid + s]);
        __syncthreads();
      }
      mx = red[0]; __syncthreads();
      float sm = 0.f;
      for (int i = tid; i < V_; i += TPB) sm += __expf(row[i] - mx);
      red[tid] = sm; __syncthreads();
      for (int s = TPB / 2; s > 0; s >>= 1) {
        if (tid < s) red[tid] += red[tid + s];
        __syncthreads();
      }
      if (tid == 0) {
        int tgt = inputs[t * B_ + wg];
        rrow_g[wg] = -(row[tgt] - mx - __logf(red[0]));
      }
    }
    gridbar(cnt, sns, &lsense, nwg);

    // ---- P12: deterministic cross-row recon sum ----
    if (wg == 0 && tid == 0) {
      float s = 0.f;
      for (int b = 0; b < B_; ++b) s += rrow_g[b];
      acc_g[1] += s * (1.f / (float)B_);
    }
    gridbar(cnt, sns, &lsense, nwg);
  }

  if (wg == 0 && tid == 0) {
    g_f32_p dout = (g_f32_p)A.dout;
    dout[0] = acc_g[0];   // sum klds
    dout[1] = acc_g[1];   // sum recons
  }
}

// ---------------- host launch ----------------
extern "C" void kernel_launch(void* const* d_in, const int* in_sizes, int n_in,
                              void* d_out, int out_size, void* d_ws, size_t ws_size,
                              hipStream_t stream) {
  (void)in_sizes; (void)n_in; (void)out_size; (void)ws_size;

  // weight shapes (N rows x K cols, row-major over K) and d_in indices,
  // matching setup_inputs() dict order.
  static const int wN[NW]  = { H_, H_, H_, H2_, H2_, Z_, Z_, H2_, Z_, Z_,
                               H2_, H2_, E_, E_, 3*H_, 3*H_, 3*H_, 3*H_, V_ };
  static const int wK[NW]  = { E_, H_, Z_, H_+H2_, H2_, H2_, H2_, H2_, H2_, H2_,
                               H_+H2_, H2_, H2_, H2_, 2*H_, H_, 2*H_, H_, H2_ };
  static const int wSrc[NW]= { 4, 6, 8, 10, 12, 14, 16, 18, 20, 22,
                               24, 26, 28, 30, 32, 33, 34, 35, 36 };
  static const int bSrc[NB]= { 5, 7, 9, 11, 13, 15, 17, 19, 21, 23, 25, 27, 29, 31, 37 };
  static const int sCnt[NS]= { B_*E_, B_*H_, B_*CATW, B_*H2_, B_*H2_, B_*Z_, B_*Z_,
                               B_*H2_, B_*Z_, B_*Z_, B_*Z_, B_*H2_, B_*H2_,
                               B_*3*H_, B_*3*H_, B_*3*H_, B_*3*H_, B_*V_, B_, 2 };

  char* ws = (char*)d_ws;
  size_t off = 0;
  auto alloc = [&](size_t bytes) -> void* {
    void* p = ws + off;
    off += (bytes + 255) & ~(size_t)255;
    return p;
  };

  PrepArgs pa;
  VArgs va;
  for (int i = 0; i < NW; ++i) {
    size_t n = (size_t)wN[i] * (size_t)wK[i];
    bf16_t* dst = (bf16_t*)alloc(n * sizeof(bf16_t));
    pa.src[i] = (const float*)d_in[wSrc[i]];
    pa.dst[i] = dst;
    pa.cnt[i] = (int)n;
    va.w[i]   = dst;
  }
  for (int i = 0; i < NS; ++i) va.s[i] = (float*)alloc((size_t)sCnt[i] * sizeof(float));
  unsigned* bar = (unsigned*)alloc(256);

  va.inputs = (const int*)d_in[0];
  va.eps    = (const float*)d_in[2];
  va.embed  = (const float*)d_in[3];
  for (int i = 0; i < NB; ++i) va.bias[i] = (const float*)d_in[bSrc[i]];
  va.bar  = bar;
  va.dout = (float*)d_out;

  pa.bar = bar;
  pa.acc = va.s[S_ACC];

  hipLaunchKernelGGL(vrnn_prep, dim3(256), dim3(256), 0, stream, pa);
  hipLaunchKernelGGL(vrnn_main, dim3(NWG), dim3(TPB), 0, stream, va);
}
